// HGTNet_51857435132408
// MI455X (gfx1250) — matrix operand order repaired and checked by reading, not compiled, not run on hardware
//
#include <hip/hip_runtime.h>
#include <math.h>

// ---------------- problem constants (match reference) ----------------
constexpr int NN   = 50000;     // nodes
constexpr int NE   = 400000;    // edges
constexpr int INDIM = 256;
constexpr int NHID  = 128;
constexpr int NT    = 3;        // node types
constexpr int NR    = 6;        // relations
constexpr int NH    = 8;        // heads
constexpr int DK    = 16;       // head dim
constexpr int NL    = 2;        // layers
constexpr float INV_SQRT_DK = 0.25f;  // 1/sqrt(16)

constexpr int ROW_TILES = NN / 16;     // 3125 (exact)
constexpr int COL_TILES = NHID / 16;   // 8
constexpr int GEMM_WAVES = ROW_TILES * COL_TILES;   // 25000
constexpr int GEMM_BLOCKS = GEMM_WAVES / 8;         // 3125 (block = 256 thr = 8 waves)

typedef __attribute__((ext_vector_type(16))) _Float16 v16h;
typedef __attribute__((ext_vector_type(8)))  float    v8f;

#define WMMA16(a, b, c) \
  __builtin_amdgcn_wmma_f32_16x16x32_f16(false, (a), false, (b), (short)0, (c), false, false)

__device__ __forceinline__ v8f v8zero() {
  v8f z = {0.f, 0.f, 0.f, 0.f, 0.f, 0.f, 0.f, 0.f};
  return z;
}

// --- fragment loaders (on-the-fly f32 -> f16 conversion) ---
// A (16x32, f16): lane l holds row l%16; elements 0..7 = K k0+(l/16)*8+e,
// elements 8..15 = K k0+16+(l/16)*8+e.
__device__ __forceinline__ v16h load_a16(const float* X, int ld, int nb, int k0, int lane) {
  int row = lane & 15, half = lane >> 4;
  const float* p = X + (size_t)(nb + row) * ld + (k0 + half * 8);
  v16h a;
#pragma unroll
  for (int e = 0; e < 8; ++e) a[e] = (_Float16)p[e];
#pragma unroll
  for (int e = 0; e < 8; ++e) a[e + 8] = (_Float16)p[16 + e];
  return a;
}

__device__ __forceinline__ float gelu_exact(float v) {
  return 0.5f * v * (1.0f + erff(v * 0.70710678118654752f));
}

__device__ __forceinline__ v16h load_a16_gelu(const float* X, int ld, int nb, int k0, int lane) {
  int row = lane & 15, half = lane >> 4;
  const float* p = X + (size_t)(nb + row) * ld + (k0 + half * 8);
  v16h a;
#pragma unroll
  for (int e = 0; e < 8; ++e) a[e] = (_Float16)gelu_exact(p[e]);
#pragma unroll
  for (int e = 0; e < 8; ++e) a[e + 8] = (_Float16)gelu_exact(p[16 + e]);
  return a;
}

// B (32x16, f16): lane l holds column c0+l%16; elements 0..7 = K k0+(l/16)*8+e,
// elements 8..15 = K k0+16+(l/16)*8+e.
__device__ __forceinline__ v16h load_b16(const float* W, int ld, int k0, int c0, int lane) {
  int half = lane >> 4;
  const float* p = W + (size_t)(k0 + half * 8) * ld + c0 + (lane & 15);
  v16h b;
#pragma unroll
  for (int e = 0; e < 8; ++e) b[e] = (_Float16)p[(size_t)e * ld];
#pragma unroll
  for (int e = 0; e < 8; ++e) b[e + 8] = (_Float16)p[(size_t)(16 + e) * ld];
  return b;
}

// ---------------- adapter: x = tanh(per_type_linear(nf, adapt_w, adapt_b)) ----------------
__global__ __launch_bounds__(256) void adapter_kernel(
    const float* __restrict__ nf, const int* __restrict__ ntype,
    const float* __restrict__ W, const float* __restrict__ B,
    float* __restrict__ xout) {
  int wave = blockIdx.x * 8 + (threadIdx.x >> 5);
  if (wave >= GEMM_WAVES) return;
  int lane = threadIdx.x & 31;
  int ct = wave & 7, rt = wave >> 3;
  int nb = rt * 16, c0 = ct * 16;

  v8f acc0 = v8zero(), acc1 = v8zero(), acc2 = v8zero();
  for (int k0 = 0; k0 < INDIM; k0 += 32) {
    v16h a  = load_a16(nf, INDIM, nb, k0, lane);
    v16h b0 = load_b16(W + (size_t)0 * INDIM * NHID, NHID, k0, c0, lane);
    acc0 = WMMA16(a, b0, acc0);
    v16h b1 = load_b16(W + (size_t)1 * INDIM * NHID, NHID, k0, c0, lane);
    acc1 = WMMA16(a, b1, acc1);
    v16h b2 = load_b16(W + (size_t)2 * INDIM * NHID, NHID, k0, c0, lane);
    acc2 = WMMA16(a, b2, acc2);
  }
  int half = lane >> 4, col = c0 + (lane & 15);
  int ty[8];
#pragma unroll
  for (int i = 0; i < 8; ++i) ty[i] = ntype[nb + i + 8 * half];
#pragma unroll
  for (int i = 0; i < 8; ++i) {
    int node = nb + i + 8 * half;
    int t = ty[i];
    float v = (t == 0) ? acc0[i] : (t == 1) ? acc1[i] : acc2[i];
    v = tanhf(v + B[t * NHID + col]);
    xout[(size_t)node * NHID + col] = v;
  }
}

// ---------------- fused per-type K/Q/V linears ----------------
__global__ __launch_bounds__(256) void kqv_kernel(
    const float* __restrict__ x, const int* __restrict__ ntype,
    const float* __restrict__ Wk, const float* __restrict__ bk,
    const float* __restrict__ Wq, const float* __restrict__ bq,
    const float* __restrict__ Wv, const float* __restrict__ bv,
    float* __restrict__ kx, float* __restrict__ qx, float* __restrict__ vx) {
  int wave = blockIdx.x * 8 + (threadIdx.x >> 5);
  if (wave >= GEMM_WAVES) return;
  int lane = threadIdx.x & 31;
  int ct = wave & 7, rt = wave >> 3;
  int nb = rt * 16, c0 = ct * 16;

  v8f acc[3][3];
#pragma unroll
  for (int w = 0; w < 3; ++w)
#pragma unroll
    for (int t = 0; t < 3; ++t) acc[w][t] = v8zero();

  const float* Ws[3] = {Wk, Wq, Wv};
  for (int k0 = 0; k0 < NHID; k0 += 32) {
    v16h a = load_a16(x, NHID, nb, k0, lane);
#pragma unroll
    for (int w = 0; w < 3; ++w)
#pragma unroll
      for (int t = 0; t < 3; ++t) {
        v16h b = load_b16(Ws[w] + (size_t)t * NHID * NHID, NHID, k0, c0, lane);
        acc[w][t] = WMMA16(a, b, acc[w][t]);
      }
  }
  int half = lane >> 4, col = c0 + (lane & 15);
  int ty[8];
#pragma unroll
  for (int i = 0; i < 8; ++i) ty[i] = ntype[nb + i + 8 * half];

  float* Ys[3] = {kx, qx, vx};
  const float* Bs[3] = {bk, bq, bv};
#pragma unroll
  for (int w = 0; w < 3; ++w)
#pragma unroll
    for (int i = 0; i < 8; ++i) {
      int node = nb + i + 8 * half;
      int t = ty[i];
      float v = (t == 0) ? acc[w][0][i] : (t == 1) ? acc[w][1][i] : acc[w][2][i];
      Ys[w][(size_t)node * NHID + col] = v + Bs[w][t * NHID + col];
    }
}

// ---------------- per-layer init: agg=0, max=-inf, sum=0 ----------------
__global__ __launch_bounds__(256) void init_kernel(
    float* __restrict__ mx, float* __restrict__ ssum, float* __restrict__ agg) {
  int i = blockIdx.x * blockDim.x + threadIdx.x;
  if (i < NN * NHID) agg[i] = 0.f;
  if (i < NN * NH) {
    mx[i] = -INFINITY;
    ssum[i] = 0.f;
  }
}

// float atomic max via signed/unsigned int ordering trick
__device__ __forceinline__ void atomicMaxF(float* addr, float val) {
  if (val >= 0.0f)
    atomicMax((int*)addr, __float_as_int(val));
  else
    atomicMin((unsigned int*)addr, __float_as_uint(val));
}

// ---------------- edge pass 1: score + segment max ----------------
__global__ __launch_bounds__(256) void edge_score_kernel(
    const int* __restrict__ eidx, const int* __restrict__ etype,
    const float* __restrict__ kx, const float* __restrict__ qx,
    const float* __restrict__ ratt, const float* __restrict__ pri,
    float* __restrict__ att, float* __restrict__ mx) {
  int idx = blockIdx.x * blockDim.x + threadIdx.x;
  if (idx >= NE * NH) return;
  int e = idx >> 3, h = idx & 7;
  int s = eidx[e], d = eidx[NE + e], r = etype[e];
  const float* kv = kx + (size_t)s * NHID + h * DK;
  const float* qv = qx + (size_t)d * NHID + h * DK;
  const float* m  = ratt + (size_t)(r * NH + h) * DK * DK;
  float sc = 0.f;
#pragma unroll
  for (int dd = 0; dd < DK; ++dd) {
    float ke = 0.f;
#pragma unroll
    for (int k = 0; k < DK; ++k) ke = fmaf(kv[k], m[k * DK + dd], ke);
    sc = fmaf(qv[dd], ke, sc);
  }
  sc = sc * pri[r * NH + h] * INV_SQRT_DK;
  att[idx] = sc;
  atomicMaxF(&mx[d * NH + h], sc);
}

// ---------------- edge pass 2: exp + segment sum ----------------
__global__ __launch_bounds__(256) void edge_expsum_kernel(
    const int* __restrict__ eidx, float* __restrict__ att,
    const float* __restrict__ mx, float* __restrict__ ssum) {
  int idx = blockIdx.x * blockDim.x + threadIdx.x;
  if (idx >= NE * NH) return;
  int e = idx >> 3, h = idx & 7;
  int d = eidx[NE + e];
  float w = __expf(att[idx] - mx[d * NH + h]);
  att[idx] = w;
  atomicAdd(&ssum[d * NH + h], w);
}

// ---------------- edge pass 3: message + attention-weighted scatter ----------------
__global__ __launch_bounds__(256) void edge_scatter_kernel(
    const int* __restrict__ eidx, const int* __restrict__ etype,
    const float* __restrict__ vx, const float* __restrict__ rmsg,
    const float* __restrict__ att, const float* __restrict__ ssum,
    float* __restrict__ agg) {
  int idx = blockIdx.x * blockDim.x + threadIdx.x;
  if (idx >= NE * NH) return;
  int e = idx >> 3, h = idx & 7;
  int s = eidx[e], d = eidx[NE + e], r = etype[e];
  const float* vv = vx + (size_t)s * NHID + h * DK;
  const float* m  = rmsg + (size_t)(r * NH + h) * DK * DK;
  float a = att[idx] / (ssum[d * NH + h] + 1e-16f);
  float* dst = agg + (size_t)d * NHID + h * DK;
#pragma unroll
  for (int dd = 0; dd < DK; ++dd) {
    float mv = 0.f;
#pragma unroll
    for (int k = 0; k < DK; ++k) mv = fmaf(vv[k], m[k * DK + dd], mv);
    atomicAdd(&dst[dd], mv * a);
  }
}

// ---------------- update: gelu -> per-type Wa -> sigmoid-gated residual ----------------
__global__ __launch_bounds__(256) void update_kernel(
    const float* __restrict__ agg, const float* __restrict__ xin,
    const int* __restrict__ ntype, const float* __restrict__ Wa,
    const float* __restrict__ ba, const float* __restrict__ skip,
    float* __restrict__ xout) {
  int wave = blockIdx.x * 8 + (threadIdx.x >> 5);
  if (wave >= GEMM_WAVES) return;
  int lane = threadIdx.x & 31;
  int ct = wave & 7, rt = wave >> 3;
  int nb = rt * 16, c0 = ct * 16;

  v8f acc0 = v8zero(), acc1 = v8zero(), acc2 = v8zero();
  for (int k0 = 0; k0 < NHID; k0 += 32) {
    v16h a  = load_a16_gelu(agg, NHID, nb, k0, lane);
    v16h b0 = load_b16(Wa + (size_t)0 * NHID * NHID, NHID, k0, c0, lane);
    acc0 = WMMA16(a, b0, acc0);
    v16h b1 = load_b16(Wa + (size_t)1 * NHID * NHID, NHID, k0, c0, lane);
    acc1 = WMMA16(a, b1, acc1);
    v16h b2 = load_b16(Wa + (size_t)2 * NHID * NHID, NHID, k0, c0, lane);
    acc2 = WMMA16(a, b2, acc2);
  }
  int half = lane >> 4, col = c0 + (lane & 15);
#pragma unroll
  for (int i = 0; i < 8; ++i) {
    int node = nb + i + 8 * half;
    int t = ntype[node];
    float trans = ((t == 0) ? acc0[i] : (t == 1) ? acc1[i] : acc2[i]) + ba[t * NHID + col];
    float alpha = 1.0f / (1.0f + __expf(-skip[t]));
    float xi = xin[(size_t)node * NHID + col];
    xout[(size_t)node * NHID + col] = trans * alpha + xi * (1.0f - alpha);
  }
}

// ---------------- host orchestration ----------------
extern "C" void kernel_launch(void* const* d_in, const int* in_sizes, int n_in,
                              void* d_out, int out_size, void* d_ws, size_t ws_size,
                              hipStream_t stream) {
  (void)in_sizes; (void)n_in; (void)out_size; (void)ws_size;
  const float* nf      = (const float*)d_in[0];
  const int*   ntype   = (const int*)d_in[1];
  const int*   eidx    = (const int*)d_in[2];
  const int*   etype   = (const int*)d_in[3];
  const float* adapt_w = (const float*)d_in[4];
  const float* adapt_b = (const float*)d_in[5];
  const float* Wk      = (const float*)d_in[6];
  const float* bk      = (const float*)d_in[7];
  const float* Wq      = (const float*)d_in[8];
  const float* bq      = (const float*)d_in[9];
  const float* Wv      = (const float*)d_in[10];
  const float* bv      = (const float*)d_in[11];
  const float* Wa      = (const float*)d_in[12];
  const float* ba      = (const float*)d_in[13];
  const float* rel_pri = (const float*)d_in[14];
  const float* rel_att = (const float*)d_in[15];
  const float* rel_msg = (const float*)d_in[16];
  const float* skip    = (const float*)d_in[17];

  float* ws = (float*)d_ws;
  const size_t NX = (size_t)NN * NHID;           // 6.4M
  float* x0   = ws;
  float* x1   = ws + 1 * NX;
  float* kx   = ws + 2 * NX;
  float* qx   = ws + 3 * NX;
  float* vx   = ws + 4 * NX;
  float* agg  = ws + 5 * NX;
  float* att  = ws + 6 * NX;                     // NE*NH = 3.2M
  float* mx   = att + (size_t)NE * NH;           // NN*NH
  float* ssum = mx + (size_t)NN * NH;            // NN*NH

  const int edgeBlocks = (NE * NH + 255) / 256;  // 12500
  const int initBlocks = (NN * NHID + 255) / 256;

  adapter_kernel<<<GEMM_BLOCKS, 256, 0, stream>>>(nf, ntype, adapt_w, adapt_b, x0);

  const float* xcur = x0;
  for (int l = 0; l < NL; ++l) {
    float* xnext = (l == NL - 1) ? (float*)d_out : x1;
    const size_t wOff = (size_t)l * NT * NHID * NHID;
    const size_t bOff = (size_t)l * NT * NHID;
    const size_t rOff = (size_t)l * NR * NH * DK * DK;

    init_kernel<<<initBlocks, 256, 0, stream>>>(mx, ssum, agg);
    kqv_kernel<<<GEMM_BLOCKS, 256, 0, stream>>>(
        xcur, ntype, Wk + wOff, bk + bOff, Wq + wOff, bq + bOff,
        Wv + wOff, bv + bOff, kx, qx, vx);
    edge_score_kernel<<<edgeBlocks, 256, 0, stream>>>(
        eidx, etype, kx, qx, rel_att + rOff, rel_pri + (size_t)l * NR * NH, att, mx);
    edge_expsum_kernel<<<edgeBlocks, 256, 0, stream>>>(eidx, att, mx, ssum);
    edge_scatter_kernel<<<edgeBlocks, 256, 0, stream>>>(
        eidx, etype, vx, rel_msg + rOff, att, ssum, agg);
    update_kernel<<<GEMM_BLOCKS, 256, 0, stream>>>(
        agg, xcur, ntype, Wa + wOff, ba + bOff, skip + (size_t)l * NT, xnext);
    xcur = xnext;
  }
}
